// DecoderLSTM_90847148245028
// MI455X (gfx1250) — matrix operand order repaired and checked by reading, not compiled, not run on hardware
//
#include <hip/hip_runtime.h>
#include <hip/hip_bf16.h>

typedef __bf16 bf16_t;
typedef __attribute__((ext_vector_type(16))) __bf16 v16bf;
typedef __attribute__((ext_vector_type(8)))  __bf16 v8bf;
typedef __attribute__((ext_vector_type(8)))  float  v8f;

// Model dims
#define Bn 256
#define Tn 512
#define Hn 512
#define En 128
#define NCn 27
#define G4H 2048   // 4*H

// ---------------------------------------------------------------------------
// Fragment loaders: this code's own contiguous k order, shared by A and B
// A 16x32 bf16: lane l -> row (l&15), 16 contiguous K values starting kb
//               with kb = (l>>4)*16
// B 32x16 bf16: lane l -> col (l&15), 16 contiguous K values starting (l>>4)*16
// ---------------------------------------------------------------------------
__device__ __forceinline__ v16bf load_a_frag(const bf16_t* __restrict__ base,
                                             int rowStride, int mBase, int k0,
                                             int lane) {
  int row = mBase + (lane & 15);
  int kb  = (lane >> 4) * 16;
  const bf16_t* p = base + (size_t)row * rowStride + k0 + kb;
  v8bf lo = *(const v8bf*)p;
  v8bf hi = *(const v8bf*)(p + 8);
  v16bf a;
#pragma unroll
  for (int i = 0; i < 8; ++i) { a[i] = lo[i]; a[i + 8] = hi[i]; }
  return a;
}

__device__ __forceinline__ v16bf load_b_frag(const bf16_t* __restrict__ w,
                                             int kStride, int colBase, int k0,
                                             int lane) {
  int g  = colBase + (lane & 15);
  int kb = (lane >> 4) * 16;
  return *(const v16bf*)(w + (size_t)g * kStride + k0 + kb);
}

// Fully-unrolled K loop: KLEN is a compile-time constant so the whole chain
// becomes SSA (no cross-iteration register copies, no WMMA hazard NOPs) and
// the scheduler hoists load clauses across WMMA groups for latency overlap.
template <int KLEN>
__device__ __forceinline__ void gemm_phase(v8f* acc,
                                           const bf16_t* __restrict__ X,
                                           int xStride, int mBase,
                                           const bf16_t* __restrict__ W,
                                           int kStride, int colBase,
                                           int lane) {
#pragma unroll
  for (int k0 = 0; k0 < KLEN; k0 += 32) {
    v16bf a = load_a_frag(X, xStride, mBase, k0, lane);
    v16bf b[4];
#pragma unroll
    for (int t = 0; t < 4; ++t)
      b[t] = load_b_frag(W, kStride, colBase + t * 16, k0, lane);
#pragma unroll
    for (int t = 0; t < 4; ++t)
      acc[t] = __builtin_amdgcn_wmma_f32_16x16x32_bf16(
          false, a, false, b[t], (short)0, acc[t], false, false);
  }
}

__device__ __forceinline__ float sigf(float x) {
  return 1.0f / (1.0f + __expf(-x));
}

#define LDSF(g, m, n) lds[(((g)*32 + (m)) * 66) + (n)]

// ---------------------------------------------------------------------------
// One LSTM layer (both directions) for one timestep, executed by 128
// workgroups (blk = dir*64 + mBlk*8 + hChunk), 256 threads / 8 waves each.
// Waves split 2(M) x 4(gate) so the workgroup owns all four gates for
// 32 batch rows x 64 hidden units and finishes the elementwise in-kernel.
//   outInit : pre-fill out[:,t,:] with bias / mask-zero (L0 role)
//   doLogit : partial output projection via f32 atomics (L1 role)
// ---------------------------------------------------------------------------
template <int KX>
__device__ __forceinline__ void layer_body(
    float* lds, int blk, int tid,
    const bf16_t* __restrict__ X, int xStride,
    const bf16_t* __restrict__ Hprev,               // [B][1024], dir cols
    const bf16_t* __restrict__ Wih,                 // [2][2048][KX]
    const bf16_t* __restrict__ Whh,                 // [2][2048][512]
    const float* __restrict__ bias,                 // [2][2048]
    float* __restrict__ Cst,                        // [2][B][512]
    bf16_t* __restrict__ Hout,                      // [B][1024]
    int step, const int* __restrict__ lens,
    const float* __restrict__ bout, float* __restrict__ out,
    bool outInit, bool doLogit, const float* __restrict__ Wout) {
  const int dir    = blk >> 6;
  const int rest   = blk & 63;
  const int mBlk   = rest >> 3;
  const int hChunk = rest & 7;
  const int wave = tid >> 5;
  const int lane = tid & 31;
  const int mw   = wave & 1;
  const int gate = wave >> 1;                    // 0:i 1:f 2:g 3:o
  const int mBase   = mBlk * 32 + mw * 16;
  const int colBase = gate * 512 + hChunk * 64;  // gate column in [0,2048)

  // L0-role: initialize this step's logit slots (bias if active, else 0)
  // before L1-role atomics of the SAME step arrive in the next launch.
  if (outInit && dir == 0 && hChunk == 0) {
    for (int idx = tid; idx < 32 * NCn; idx += 256) {
      int m = idx / NCn, c = idx - m * NCn;
      int brow = mBlk * 32 + m;
      out[((size_t)brow * Tn + step) * NCn + c] =
          (step < lens[brow]) ? bout[c] : 0.0f;
    }
  }

  const bf16_t* WihD = Wih + (size_t)dir * G4H * KX;
  const bf16_t* WhhD = Whh + (size_t)dir * G4H * Hn;

  v8f acc[4];
#pragma unroll
  for (int t = 0; t < 4; ++t)
#pragma unroll
    for (int r = 0; r < 8; ++r) acc[t][r] = 0.0f;

  // Phase 1: X @ Wih^T (K = KX); Phase 2: Hprev(dir) @ Whh^T (K = 512)
  gemm_phase<KX>(acc, X, xStride, mBase, WihD, KX, colBase, lane);
  gemm_phase<Hn>(acc, Hprev + dir * Hn, 2 * Hn, mBase, WhhD, Hn, colBase,
                 lane);

  // Scatter gate tiles to LDS: C layout lane l, vgpr r ->
  // (m = r + 8*(l>>4), n = l&15) of each 16x16 tile.
#pragma unroll
  for (int t = 0; t < 4; ++t)
#pragma unroll
    for (int r = 0; r < 8; ++r) {
      int m = mw * 16 + r + 8 * (lane >> 4);
      int n = t * 16 + (lane & 15);
      LDSF(gate, m, n) = acc[t][r];
    }
  __syncthreads();

  // Elementwise LSTM update: each (m,j) pair owned by exactly one thread,
  // which may safely overwrite LDSF(0,m,j) with h2 for the logit phase.
  const float* biasD = bias + (size_t)dir * G4H;
#pragma unroll
  for (int p = 0; p < 8; ++p) {
    int idx = tid + p * 256;
    int m = idx >> 6;
    int j = idx & 63;
    int brow = mBlk * 32 + m;
    int hid  = hChunk * 64 + j;
    float gi = LDSF(0, m, j) + biasD[hid];
    float gf = LDSF(1, m, j) + biasD[512 + hid];
    float gg = LDSF(2, m, j) + biasD[1024 + hid];
    float go = LDSF(3, m, j) + biasD[1536 + hid];
    size_t ci = ((size_t)dir * Bn + brow) * Hn + hid;
    float c  = Cst[ci];
    float c2 = sigf(gf) * c + sigf(gi) * tanhf(gg);
    float h2 = sigf(go) * tanhf(c2);
    Cst[ci] = c2;
    Hout[(size_t)brow * (2 * Hn) + dir * Hn + hid] = (bf16_t)h2;
    LDSF(0, m, j) = h2;
  }

  // L1-role: partial output projection. This WG owns y1 chunk
  // [32 rows][64 units] of one dir -> add 64-length partial dots into
  // out[b][step][c] (initialized with bias by L0-role of this step).
  if (doLogit) {
    __syncthreads();
    for (int idx = tid; idx < 32 * NCn; idx += 256) {
      int m = idx / NCn, c = idx - m * NCn;
      int brow = mBlk * 32 + m;
      if (step < lens[brow]) {
        const float* w = Wout + (size_t)c * (2 * Hn) + dir * Hn + hChunk * 64;
        float pacc = 0.0f;
#pragma unroll
        for (int j = 0; j < 64; ++j) pacc += LDSF(0, m, j) * w[j];
        atomicAdd(&out[((size_t)brow * Tn + step) * NCn + c], pacc);
      }
    }
  }
}

// ---------------------------------------------------------------------------
// Fused step kernel: blocks [0,128) run layer0 of step t, blocks [128,256)
// run layer1 of step t-1 (independent thanks to double-buffered h state).
// ---------------------------------------------------------------------------
struct StepArgs {
  const bf16_t* xemb;
  const bf16_t* wih0;
  const bf16_t* wih1;
  const bf16_t* whh0;
  const bf16_t* whh1;
  const float*  bias0;
  const float*  bias1;
  float*        cst0;
  float*        cst1;
  bf16_t*       hL0[2];
  bf16_t*       hL1[2];
  const float*  Wout;
  const float*  bout;
  const int*    lens;
  float*        out;
  int           t;
};

__global__ __launch_bounds__(256) void fused_step_kernel(StepArgs A) {
  __shared__ float lds[4 * 32 * 66];
  const int role = blockIdx.x >> 7;
  const int blk  = blockIdx.x & 127;
  const int tid  = threadIdx.x;

  if (role == 0) {
    if (A.t >= Tn) return;
    const int t = A.t;
    layer_body<En>(lds, blk, tid,
                   A.xemb + (size_t)t * Bn * En, En,
                   A.hL0[t & 1], A.wih0, A.whh0, A.bias0, A.cst0,
                   A.hL0[(t + 1) & 1],
                   t, A.lens, A.bout, A.out,
                   /*outInit=*/true, /*doLogit=*/false, nullptr);
  } else {
    if (A.t < 1) return;
    const int s = A.t - 1;
    layer_body<2 * Hn>(lds, blk, tid,
                       A.hL0[(s + 1) & 1], 2 * Hn,
                       A.hL1[s & 1], A.wih1, A.whh1, A.bias1, A.cst1,
                       A.hL1[(s + 1) & 1],
                       s, A.lens, A.bout, A.out,
                       /*outInit=*/false, /*doLogit=*/true, A.Wout);
  }
}

// ---------------------------------------------------------------------------
// Prep kernels
// ---------------------------------------------------------------------------
__global__ void cvt_f32_bf16_kernel(const float* __restrict__ s,
                                    bf16_t* __restrict__ d, int n) {
  int i = blockIdx.x * blockDim.x + threadIdx.x;
  if (i < n) d[i] = (bf16_t)s[i];
}

__global__ void bias_sum_kernel(const float* __restrict__ a,
                                const float* __restrict__ b,
                                float* __restrict__ d, int n) {
  int i = blockIdx.x * blockDim.x + threadIdx.x;
  if (i < n) d[i] = a[i] + b[i];
}

__global__ void copy_f32_kernel(const float* __restrict__ s,
                                float* __restrict__ d, int n) {
  int i = blockIdx.x * blockDim.x + threadIdx.x;
  if (i < n) d[i] = s[i];
}

__global__ void copy_i32_kernel(const int* __restrict__ s,
                                int* __restrict__ d, int n) {
  int i = blockIdx.x * blockDim.x + threadIdx.x;
  if (i < n) d[i] = s[i];
}

// x_bf16[t][b][e] = embed[tgt[b][t]][e]
__global__ void embed_kernel(const int* __restrict__ tgt,
                             const float* __restrict__ emb,
                             bf16_t* __restrict__ x) {
  int i = blockIdx.x * blockDim.x + threadIdx.x;  // over T*B*E
  int t = i / (Bn * En);
  int r = i - t * (Bn * En);
  int b = r / En;
  int e = r - b * En;
  int tok = tgt[(size_t)b * Tn + t];
  x[i] = (bf16_t)emb[(size_t)tok * En + e];
}

// h0[4][B][512] -> hL0[0] / hL1[0] bf16 buffers ([B][1024], dir cols)
__global__ void h_init_kernel(const float* __restrict__ h0,
                              bf16_t* __restrict__ hL0,
                              bf16_t* __restrict__ hL1) {
  int i = blockIdx.x * blockDim.x + threadIdx.x;  // over 4*B*512
  if (i >= 4 * Bn * Hn) return;
  int s = i / (Bn * Hn);  // layer*2+dir
  int r = i - s * (Bn * Hn);
  int b = r / Hn;
  int j = r - b * Hn;
  bf16_t v = (bf16_t)h0[i];
  int dir = s & 1;
  bf16_t* dst = (s < 2) ? hL0 : hL1;
  dst[(size_t)b * (2 * Hn) + dir * Hn + j] = v;
}

// ---------------------------------------------------------------------------
extern "C" void kernel_launch(void* const* d_in, const int* in_sizes, int n_in,
                              void* d_out, int out_size, void* d_ws,
                              size_t ws_size, hipStream_t stream) {
  const int*   tgt   = (const int*)d_in[0];
  const int*   lens  = (const int*)d_in[1];
  const float* h0    = (const float*)d_in[2];
  const float* c0    = (const float*)d_in[3];
  const float* embed = (const float*)d_in[4];
  const float* Wih0  = (const float*)d_in[5];
  const float* Wih1  = (const float*)d_in[6];
  const float* Whh   = (const float*)d_in[7];
  const float* bih   = (const float*)d_in[8];
  const float* bhh   = (const float*)d_in[9];
  const float* Wout  = (const float*)d_in[10];
  const float* bout  = (const float*)d_in[11];
  float* out = (float*)d_out;

  // Workspace carve-up (256B aligned)
  char* ws = (char*)d_ws;
  size_t off = 0;
  auto carve = [&](size_t bytes) -> char* {
    char* p = ws + off;
    off = (off + bytes + 255) & ~(size_t)255;
    return p;
  };
  bf16_t* xemb  = (bf16_t*)carve((size_t)Tn * Bn * En * 2);        // 33.5 MB
  bf16_t* wih0b = (bf16_t*)carve((size_t)2 * G4H * En * 2);
  bf16_t* wih1b = (bf16_t*)carve((size_t)2 * G4H * (2 * Hn) * 2);
  bf16_t* whhb  = (bf16_t*)carve((size_t)4 * G4H * Hn * 2);
  float*  biasb = (float*)carve((size_t)4 * G4H * 4);
  float*  cst   = (float*)carve((size_t)4 * Bn * Hn * 4);
  bf16_t* hL0a  = (bf16_t*)carve((size_t)Bn * 2 * Hn * 2);
  bf16_t* hL0b  = (bf16_t*)carve((size_t)Bn * 2 * Hn * 2);
  bf16_t* hL1a  = (bf16_t*)carve((size_t)Bn * 2 * Hn * 2);
  bf16_t* hL1b  = (bf16_t*)carve((size_t)Bn * 2 * Hn * 2);
  (void)ws_size;

  // ---- prep ----
  {
    int n = 2 * G4H * En;
    cvt_f32_bf16_kernel<<<(n + 255) / 256, 256, 0, stream>>>(Wih0, wih0b, n);
  }
  {
    int n = 2 * G4H * 2 * Hn;
    cvt_f32_bf16_kernel<<<(n + 255) / 256, 256, 0, stream>>>(Wih1, wih1b, n);
  }
  {
    int n = 4 * G4H * Hn;
    cvt_f32_bf16_kernel<<<(n + 255) / 256, 256, 0, stream>>>(Whh, whhb, n);
  }
  {
    int n = 4 * G4H;
    bias_sum_kernel<<<(n + 255) / 256, 256, 0, stream>>>(bih, bhh, biasb, n);
  }
  {
    int n = 4 * Bn * Hn;
    copy_f32_kernel<<<(n + 255) / 256, 256, 0, stream>>>(c0, cst, n);
    h_init_kernel<<<(n + 255) / 256, 256, 0, stream>>>(h0, hL0a, hL1a);
  }
  {
    int n = Tn * Bn * En;
    embed_kernel<<<n / 256, 256, 0, stream>>>(tgt, embed, xemb);
  }
  // target_lengths appended to d_out (int32 bit pattern)
  copy_i32_kernel<<<1, 256, 0, stream>>>(lens,
                                         (int*)out + (size_t)Bn * Tn * NCn, Bn);

  // ---- pipelined timesteps: launch t runs L0(t) || L1(t-1) ----
  StepArgs A;
  A.xemb  = xemb;
  A.wih0  = wih0b;
  A.wih1  = wih1b;
  A.whh0  = whhb;
  A.whh1  = whhb + (size_t)2 * G4H * Hn;
  A.bias0 = biasb;
  A.bias1 = biasb + (size_t)2 * G4H;
  A.cst0  = cst;
  A.cst1  = cst + (size_t)2 * Bn * Hn;
  A.hL0[0] = hL0a; A.hL0[1] = hL0b;
  A.hL1[0] = hL1a; A.hL1[1] = hL1b;
  A.Wout  = Wout;
  A.bout  = bout;
  A.lens  = lens;
  A.out   = out;

  for (int t = 0; t <= Tn; ++t) {
    A.t = t;
    fused_step_kernel<<<256, 256, 0, stream>>>(A);
  }
}
